// Seq2Seq_15393162789582
// MI455X (gfx1250) — matrix operand order repaired and checked by reading, not compiled, not run on hardware
//
#include <hip/hip_runtime.h>
#include <hip/hip_bf16.h>

// ---------------------------------------------------------------------------
// Seq2Seq LSTM for MI455X (gfx1250, wave32, WMMA).
//   Kernel 1: embedding gather + x-projection GEMM (WMMA f16 -> f32),
//             bias folded in, writes gates_x[S*B, 4H] f32 to workspace.
//   Kernel 2: encoder recurrence, single workgroup (8 waves) on one WGP.
//             h state kept as a zero-padded 48x64 f16 image in LDS so each
//             WMMA A-fragment is exactly 2x ds_load_b128 per lane.
//   Kernel 3: decoder (40 autoregressive steps), same structure.
// ---------------------------------------------------------------------------

typedef __attribute__((ext_vector_type(16))) _Float16 v16h;
typedef __attribute__((ext_vector_type(8)))  _Float16 v8h;
typedef __attribute__((ext_vector_type(4)))  _Float16 v4h;
typedef __attribute__((ext_vector_type(8)))  float    v8f;

#define SEQLEN 8192
#define NBATCH 40
#define NEMB   64
#define NHID   40
#define NGATE  160  // 4*NHID
#define LDH    64   // padded LDS row stride (halfs) for h/x images
#define NROWP  48   // padded row count (3 M-tiles x 16)

// ---- fragment builders (layouts per CDNA5 ISA §7.12.2) ----------

// 16-bit A 16x32 fragment, row-major source with ld (halfs) stride.
// Element j -> K = kbase + kh*8 + j (j<8), kbase + 16 + kh*8 + (j-8) (j>=8):
// two contiguous 8-half (128-bit) runs -> 2x ds_load_b128.
__device__ inline v16h load_a_frag(const _Float16* __restrict__ src, int row,
                                   int ldh, int kbase, int kh) {
  const _Float16* base = src + row * ldh + kbase + kh * 8;
  v8h lo = *(const v8h*)(base);
  v8h hi = *(const v8h*)(base + 16);
  return __builtin_shufflevector(lo, hi, 0, 1, 2, 3, 4, 5, 6, 7, 8, 9, 10, 11,
                                 12, 13, 14, 15);
}

// 16-bit B matrix 32x16: lane = N col (lane&15); element j -> K = kbase + kh*8 + j
// (j<8), kbase + 16 + kh*8 + (j-8) (j>=8), matching the A fragment.
// B[k][n] = W[n][k] where W is the torch-style (4H, K) weight (row-major).
__device__ inline v16h make_b_frag(const float* __restrict__ W, int Kdim, int n,
                                   int kbase, int kh) {
  v16h b;
#pragma unroll
  for (int j = 0; j < 16; ++j) {
    int k = kbase + (kh << 3) + (j < 8 ? j : j + 8);
    float v = (k < Kdim) ? W[n * Kdim + k] : 0.0f;
    b[j] = (_Float16)v;
  }
  return b;
}

__device__ inline v8f wmma_acc(v16h a, v16h b, v8f c) {
  return __builtin_amdgcn_wmma_f32_16x16x32_f16(false, a, false, b, (short)0, c,
                                                false, false);
}

__device__ inline float sigmf(float x) { return 1.0f / (1.0f + __expf(-x)); }
// tanh(x) = 2*sigmoid(2x) - 1  (saturates cleanly, no inf/inf NaN)
__device__ inline float tanh_fast(float x) { return 2.0f * sigmf(2.0f * x) - 1.0f; }

// ---------------------------------------------------------------------------
// Kernel 1: gates_x[m, n] = emb[source[m]] @ enc_Wih^T + (bih + bhh)
//   m = t*40 + b in [0, 327680), n in [0, 160).  M tiles of 16, 8 per block.
// ---------------------------------------------------------------------------
__global__ __launch_bounds__(256) void xproj_kernel(
    const int* __restrict__ source, const float* __restrict__ emb,
    const float* __restrict__ Wih, const float* __restrict__ bih,
    const float* __restrict__ bhh, float* __restrict__ gates_x) {
  __shared__ alignas(16) _Float16 arows[8][16 * NEMB];  // per-wave staging, f16

  const int tid   = threadIdx.x;
  const int wave  = tid >> 5;
  const int lane  = tid & 31;
  const int lr    = lane & 15;
  const int kh    = lane >> 4;
  const int mtile = blockIdx.x * 8 + wave;  // < 20480

  // stage: each lane loads half of one embedded row (32 f32 -> f16 LDS)
  {
    int r  = lane >> 1;
    int cb = (lane & 1) * 32;
    int m  = mtile * 16 + r;
    const float* er = emb + (size_t)source[m] * NEMB + cb;
    _Float16* dst = &arows[wave][r * NEMB + cb];
#pragma unroll
    for (int c = 0; c < 32; c += 4) {
      float4 v = *(const float4*)(er + c);
      v4h pk;
      pk[0] = (_Float16)v.x;
      pk[1] = (_Float16)v.y;
      pk[2] = (_Float16)v.z;
      pk[3] = (_Float16)v.w;
      *(v4h*)(dst + c) = pk;  // ds_store_b64
    }
  }
  __syncthreads();

  v16h a0 = load_a_frag(&arows[wave][0], lr, NEMB, 0, kh);
  v16h a1 = load_a_frag(&arows[wave][0], lr, NEMB, 32, kh);

#pragma unroll
  for (int nt = 0; nt < 10; ++nt) {
    int n = nt * 16 + lr;
    v16h b0 = make_b_frag(Wih, NEMB, n, 0,  kh);
    v16h b1 = make_b_frag(Wih, NEMB, n, 32, kh);
    v8f acc = {};
    acc = wmma_acc(a0, b0, acc);
    acc = wmma_acc(a1, b1, acc);
    float bias = bih[n] + bhh[n];
#pragma unroll
    for (int r = 0; r < 8; ++r) {
      int m = mtile * 16 + r + 8 * kh;   // C layout: M = r + 8*(lane>=16)
      gates_x[(size_t)m * NGATE + n] = acc[r] + bias;
    }
  }
}

// ---------------------------------------------------------------------------
// Kernel 2: encoder recurrence. Single workgroup, 8 waves.
//   30 output tiles (mt 0..2 x nt 0..9); wave w owns tiles w, w+8, w+16, w+24
//   (clamped to 29 so control flow is uniform -> EXEC all-ones at WMMA).
// ---------------------------------------------------------------------------
__global__ __launch_bounds__(256) void enc_kernel(
    const float* __restrict__ gates_x, const float* __restrict__ Whh,
    float* __restrict__ h_out, float* __restrict__ c_out) {
  __shared__ alignas(16) _Float16 hbuf[NROWP * LDH];  // padded h image, f16
  __shared__ float cbuf[NBATCH * NHID];               // c state, f32
  __shared__ float gbuf[NBATCH * NGATE];              // raw gates for one step

  const int tid  = threadIdx.x;
  const int wave = tid >> 5;
  const int lane = tid & 31;
  const int lr   = lane & 15;
  const int kh   = lane >> 4;

  // zero the whole padded image once; pad region stays zero forever
  for (int i = tid; i < NROWP * LDH; i += 256) hbuf[i] = (_Float16)0.0f;
  for (int i = tid; i < NBATCH * NHID; i += 256) cbuf[i] = 0.0f;
  __syncthreads();

  int t_mt[4], t_nt[4];
#pragma unroll
  for (int i = 0; i < 4; ++i) {
    int tile = wave + 8 * i;
    if (tile > 29) tile = 29;  // duplicate work, identical values -> benign
    t_mt[i] = tile / 10;
    t_nt[i] = tile % 10;
  }

  // preload Whh B-fragments (K = 40, zero-padded to 64)
  v16h bf0[4], bf1[4];
#pragma unroll
  for (int i = 0; i < 4; ++i) {
    int n = t_nt[i] * 16 + lr;
    bf0[i] = make_b_frag(Whh, NHID, n, 0,  kh);
    bf1[i] = make_b_frag(Whh, NHID, n, 32, kh);
  }

  // prefetch gates_x for t = 0 into registers
  float gx[4][8];
#pragma unroll
  for (int i = 0; i < 4; ++i)
#pragma unroll
    for (int r = 0; r < 8; ++r) {
      int m = t_mt[i] * 16 + r + 8 * kh;
      gx[i][r] = (m < NBATCH)
                     ? gates_x[(size_t)m * NGATE + t_nt[i] * 16 + lr]
                     : 0.0f;
    }

  for (int t = 0; t < SEQLEN; ++t) {
    // ---- phase 1: gates = h @ Whh^T + gates_x[t]  -> gbuf ----
#pragma unroll
    for (int i = 0; i < 4; ++i) {
      const int mt = t_mt[i], nt = t_nt[i];
      v16h a0 = load_a_frag(hbuf, mt * 16 + lr, LDH, 0,  kh);
      v16h a1 = load_a_frag(hbuf, mt * 16 + lr, LDH, 32, kh);
      v8f acc = {};
      acc = wmma_acc(a0, bf0[i], acc);
      acc = wmma_acc(a1, bf1[i], acc);
#pragma unroll
      for (int r = 0; r < 8; ++r) {
        int m = mt * 16 + r + 8 * kh;
        if (m < NBATCH) gbuf[m * NGATE + nt * 16 + lr] = acc[r] + gx[i][r];
      }
    }
    __syncthreads();

    // ---- prefetch next step's gates_x (overlaps the cell update) ----
    const int tn = (t + 1 < SEQLEN) ? (t + 1) : t;
    __builtin_prefetch(&gates_x[((size_t)tn * NBATCH) * NGATE], 0, 0);
#pragma unroll
    for (int i = 0; i < 4; ++i)
#pragma unroll
      for (int r = 0; r < 8; ++r) {
        int m = t_mt[i] * 16 + r + 8 * kh;
        if (m < NBATCH)
          gx[i][r] = gates_x[((size_t)tn * NBATCH + m) * NGATE + t_nt[i] * 16 + lr];
      }

    // ---- phase 2: LSTM cell update (torch gate order i, f, g, o) ----
    for (int idx = tid; idx < NBATCH * NHID; idx += 256) {
      int m = idx / NHID;
      int j = idx - m * NHID;
      const float* gr = &gbuf[m * NGATE];
      float cn = sigmf(gr[NHID + j]) * cbuf[idx] +
                 sigmf(gr[j]) * tanh_fast(gr[2 * NHID + j]);
      float hn = sigmf(gr[3 * NHID + j]) * tanh_fast(cn);
      cbuf[idx] = cn;
      hbuf[m * LDH + j] = (_Float16)hn;  // padded image write
    }
    __syncthreads();
  }

  for (int idx = tid; idx < NBATCH * NHID; idx += 256) {
    int m = idx / NHID;
    int j = idx - m * NHID;
    h_out[idx] = (float)hbuf[m * LDH + j];
    c_out[idx] = cbuf[idx];
  }
}

// ---------------------------------------------------------------------------
// Kernel 3: decoder, 40 autoregressive steps.
//   gates = x @ Wih^T + h @ Whh^T + b;  next carry: x = h = h_new.
// ---------------------------------------------------------------------------
__global__ __launch_bounds__(256) void dec_kernel(
    const float* __restrict__ dec_in, const float* __restrict__ Wih,
    const float* __restrict__ Whh, const float* __restrict__ bih,
    const float* __restrict__ bhh, const float* __restrict__ h_in,
    const float* __restrict__ c_in, float* __restrict__ out) {
  __shared__ alignas(16) _Float16 xbuf[NROWP * LDH];
  __shared__ alignas(16) _Float16 hbuf[NROWP * LDH];
  __shared__ float cbuf[NBATCH * NHID];
  __shared__ float gbuf[NBATCH * NGATE];

  const int tid  = threadIdx.x;
  const int wave = tid >> 5;
  const int lane = tid & 31;
  const int lr   = lane & 15;
  const int kh   = lane >> 4;

  // init padded images (pad = 0) and c state
  for (int i = tid; i < NROWP * LDH; i += 256) {
    int m = i / LDH, j = i - m * LDH;
    bool v = (m < NBATCH) && (j < NHID);
    xbuf[i] = v ? (_Float16)dec_in[m * NHID + j] : (_Float16)0.0f;
    hbuf[i] = v ? (_Float16)h_in[m * NHID + j] : (_Float16)0.0f;
  }
  for (int i = tid; i < NBATCH * NHID; i += 256) cbuf[i] = c_in[i];
  __syncthreads();

  int t_mt[4], t_nt[4];
  float bn[4];
  v16h bx0[4], bx1[4], bh0[4], bh1[4];
#pragma unroll
  for (int i = 0; i < 4; ++i) {
    int tile = wave + 8 * i;
    if (tile > 29) tile = 29;
    t_mt[i] = tile / 10;
    t_nt[i] = tile % 10;
    int n = t_nt[i] * 16 + lr;
    bn[i]  = bih[n] + bhh[n];
    bx0[i] = make_b_frag(Wih, NHID, n, 0,  kh);
    bx1[i] = make_b_frag(Wih, NHID, n, 32, kh);
    bh0[i] = make_b_frag(Whh, NHID, n, 0,  kh);
    bh1[i] = make_b_frag(Whh, NHID, n, 32, kh);
  }

  for (int t = 0; t < NBATCH; ++t) {  // T == 40 decode steps
#pragma unroll
    for (int i = 0; i < 4; ++i) {
      const int mt = t_mt[i], nt = t_nt[i];
      v16h ax0 = load_a_frag(xbuf, mt * 16 + lr, LDH, 0,  kh);
      v16h ax1 = load_a_frag(xbuf, mt * 16 + lr, LDH, 32, kh);
      v16h ah0 = load_a_frag(hbuf, mt * 16 + lr, LDH, 0,  kh);
      v16h ah1 = load_a_frag(hbuf, mt * 16 + lr, LDH, 32, kh);
      v8f acc = {};
      acc = wmma_acc(ax0, bx0[i], acc);
      acc = wmma_acc(ax1, bx1[i], acc);
      acc = wmma_acc(ah0, bh0[i], acc);
      acc = wmma_acc(ah1, bh1[i], acc);
#pragma unroll
      for (int r = 0; r < 8; ++r) {
        int m = mt * 16 + r + 8 * kh;
        if (m < NBATCH) gbuf[m * NGATE + nt * 16 + lr] = acc[r] + bn[i];
      }
    }
    __syncthreads();

    for (int idx = tid; idx < NBATCH * NHID; idx += 256) {
      int m = idx / NHID;
      int j = idx - m * NHID;
      const float* gr = &gbuf[m * NGATE];
      float cn = sigmf(gr[NHID + j]) * cbuf[idx] +
                 sigmf(gr[j]) * tanh_fast(gr[2 * NHID + j]);
      float hn = sigmf(gr[3 * NHID + j]) * tanh_fast(cn);
      cbuf[idx] = cn;
      hbuf[m * LDH + j] = (_Float16)hn;  // next h
      xbuf[m * LDH + j] = (_Float16)hn;  // next x (carry is (h, h, c))
      out[(size_t)t * (NBATCH * NHID) + idx] = hn;
    }
    __syncthreads();
  }
}

// ---------------------------------------------------------------------------
extern "C" void kernel_launch(void* const* d_in, const int* in_sizes, int n_in,
                              void* d_out, int out_size, void* d_ws, size_t ws_size,
                              hipStream_t stream) {
  (void)in_sizes; (void)n_in; (void)out_size; (void)ws_size;

  const int*   source  = (const int*)d_in[0];
  const float* dec_in  = (const float*)d_in[1];
  const float* emb     = (const float*)d_in[2];
  const float* enc_Wih = (const float*)d_in[3];
  const float* enc_Whh = (const float*)d_in[4];
  const float* enc_bih = (const float*)d_in[5];
  const float* enc_bhh = (const float*)d_in[6];
  const float* dec_Wih = (const float*)d_in[7];
  const float* dec_Whh = (const float*)d_in[8];
  const float* dec_bih = (const float*)d_in[9];
  const float* dec_bhh = (const float*)d_in[10];
  float* out = (float*)d_out;

  // workspace layout: gates_x (S*B*4H f32) | h_fin (1600 f32) | c_fin (1600 f32)
  float* gates_x = (float*)d_ws;
  float* h_fin   = gates_x + (size_t)SEQLEN * NBATCH * NGATE;
  float* c_fin   = h_fin + NBATCH * NHID;

  // 327680 rows / 16 = 20480 M-tiles, 8 tiles (waves) per 256-thread block
  xproj_kernel<<<dim3(20480 / 8), dim3(256), 0, stream>>>(
      source, emb, enc_Wih, enc_bih, enc_bhh, gates_x);

  enc_kernel<<<dim3(1), dim3(256), 0, stream>>>(gates_x, enc_Whh, h_fin, c_fin);

  dec_kernel<<<dim3(1), dim3(256), 0, stream>>>(
      dec_in, dec_Wih, dec_Whh, dec_bih, dec_bhh, h_fin, c_fin, out);
}
